// SGNN_8770323218947
// MI455X (gfx1250) — matrix operand order repaired and checked by reading, not compiled, not run on hardware
//
#include <hip/hip_runtime.h>
#include <hip/hip_bf16.h>

// ---------------------------------------------------------------------------
// SGNN forward for MI455X (gfx1250, wave32).
// Dense MLP work (~75 GFLOP of [E,~320]x[~320,128] GEMMs) runs on f16 WMMA
// (v_wmma_f32_16x16x32_f16, f32 accumulate). Edge-feature assembly (gather +
// O(3)-invariant Gram matrix) is fused into LDS in front of the WMMA loop so
// the [E,305..400] activation matrix never touches HBM. Each block covers 32
// M-rows (two 16-row sub-tiles) so every B fragment load feeds two
// independent WMMAs -> halved L2 weight traffic + pipelined matrix unit.
// Graph scatter ops use f32 atomics (bandwidth-trivial vs 23.3 TB/s).
//
// d_in leaf order assumes jax pytree flattening (dict keys sorted):
//  0 x_p  1 v_p  2 h_p  3 ei_inner  4 ei_inter  5 obj_id
//  6/7 emb.w/b  8/9 emb1  10/11 emb2
//  then local[0..1], o2p[0..1], obj[0..1], each layer:
//    edge1(w,b) edge2(w,b) fcoef(w,b) nfcoef(w,b) node1(w,b) node2(w,b)
// ---------------------------------------------------------------------------

#define NPART 20000
#define NEDGE 150000
#define NOBJ  50
#define HID   128
#define ETA   0.001f
#define LDS_PAD 8
#define MT    32          // M rows per block (2 x 16-row WMMA sub-tiles)

typedef __attribute__((ext_vector_type(16))) _Float16 v16h;
typedef __attribute__((ext_vector_type(8)))  float    v8f;

// A fragment: 16x32 f16 tile (rows mapped by lane&15). Per ISA table:
// lanes 0-15 hold K kb+0..7 and kb+16..23; lanes 16-31 hold kb+8..15, kb+24..31.
__device__ __forceinline__ v16h load_a_frag(const _Float16* base, int stride,
                                            int kb, int lane) {
  int r  = lane & 15;
  int ko = (lane >> 4) << 3;               // 0 or 8
  const _Float16* p = base + r * stride + kb + ko;
  v16h v;
#pragma unroll
  for (int i = 0; i < 8; ++i) v[i] = p[i];
#pragma unroll
  for (int i = 0; i < 8; ++i) v[8 + i] = p[16 + i];
  return v;
}

// B fragment: 32x16 f16 (WT stored row-major [n][Kp]). Same K order as A:
// lanes 0-15 hold K kb+0..7, kb+16..23 (col n=lane); lanes 16-31 hold kb+8..15, kb+24..31.
__device__ __forceinline__ v16h load_b_frag(const _Float16* WTn, int Kp,
                                            int kb, int lane) {
  int n  = lane & 15;
  int ko = (lane >> 4) << 3;               // 0 or 8
  const _Float16* p = WTn + (size_t)n * Kp + kb + ko;
  v16h v;
#pragma unroll
  for (int i = 0; i < 8; ++i) v[i] = p[i];
#pragma unroll
  for (int i = 0; i < 8; ++i) v[8 + i] = p[16 + i];
  return v;
}

__device__ __forceinline__ float silu_f(float x) {
  return x / (1.f + __expf(-x));
}

// ---------------------------------------------------------------------------
// Weight conversion: W f32 [K,N] -> WT f16 [Npad][Kp], transposed, zero-padded.
__global__ __launch_bounds__(256)
void k_convert_wt(const float* __restrict__ W, _Float16* __restrict__ WT,
                  int K, int N, int Kp, int Np) {
  int i = blockIdx.x * 256 + threadIdx.x;
  if (i >= Np * Kp) return;
  int n = i / Kp, k = i - n * Kp;
  WT[i] = (n < N && k < K) ? (_Float16)W[(size_t)k * N + n] : (_Float16)0.f;
}

// Activation f32 [M,K] -> f16 [M,Kp] zero-padded.
__global__ __launch_bounds__(256)
void k_convert_act(const float* __restrict__ A, _Float16* __restrict__ out,
                   int M, int K, int Kp) {
  int i = blockIdx.x * 256 + threadIdx.x;
  if (i >= M * Kp) return;
  int m = i / Kp, k = i - m * Kp;
  out[i] = (k < K) ? (_Float16)A[(size_t)m * K + k] : (_Float16)0.f;
}

// ---------------------------------------------------------------------------
// Generic WMMA GEMM: C[M,Nreal] = act(A[M,Kp] @ WT^T + bias) (+Res)
// A f16 row-major (lda), WT f16 [Npad][Kp]. 8 waves/block, MT rows/block,
// each wave: one 16-col tile x two 16-row sub-tiles sharing the B fragment.
__global__ __launch_bounds__(256)
void k_gemm(const _Float16* __restrict__ A, int lda,
            const _Float16* __restrict__ WT,
            const float* __restrict__ bias,
            float* __restrict__ Cf, int ldc,
            _Float16* __restrict__ Ch, int ldch,
            const float* __restrict__ Res,
            int M, int Nreal, int Kp, int silu) {
  extern __shared__ _Float16 smem[];      // MT x (Kp + LDS_PAD)
  int stride = Kp + LDS_PAD;
  int tid = threadIdx.x;
  int mbase = blockIdx.x * MT;

  unsigned* s32 = (unsigned*)smem;
  const unsigned* A32 = (const unsigned*)A;
  int Kp2 = Kp >> 1, st2 = stride >> 1, lda2 = lda >> 1;
  for (int idx = tid; idx < MT * Kp2; idx += 256) {
    int r = idx / Kp2, k2 = idx - r * Kp2;
    unsigned val = 0;
    int m = mbase + r;
    if (m < M) val = A32[(size_t)m * lda2 + k2];
    s32[r * st2 + k2] = val;
  }
  __syncthreads();

  int wave = tid >> 5, lane = tid & 31;
  int Npad = (Nreal + 15) & ~15;
  for (int nb = wave * 16; nb < Npad; nb += 128) {
    v8f acc0 = {}, acc1 = {};
    const _Float16* WTn = WT + (size_t)nb * Kp;
    for (int kb = 0; kb < Kp; kb += 32) {
      v16h b  = load_b_frag(WTn, Kp, kb, lane);
      v16h a0 = load_a_frag(smem, stride, kb, lane);
      v16h a1 = load_a_frag(smem + 16 * stride, stride, kb, lane);
      acc0 = __builtin_amdgcn_wmma_f32_16x16x32_f16(false, a0, false, b,
                                                    (short)0, acc0, false, false);
      acc1 = __builtin_amdgcn_wmma_f32_16x16x32_f16(false, a1, false, b,
                                                    (short)0, acc1, false, false);
    }
    int n = nb + (lane & 15);
    int rb = (lane >> 4) << 3;
    if (n < Nreal) {
      float bv = bias ? bias[n] : 0.f;
#pragma unroll
      for (int t = 0; t < 2; ++t) {
#pragma unroll
        for (int r = 0; r < 8; ++r) {
          int m = mbase + t * 16 + rb + r;
          if (m < M) {
            float v = (t ? acc1[r] : acc0[r]) + bv;
            if (silu) v = silu_f(v);
            if (Res) v += Res[(size_t)m * ldc + n];
            if (Cf) Cf[(size_t)m * ldc + n] = v;
            if (Ch) Ch[(size_t)m * ldch + n] = (_Float16)v;
          }
        }
      }
    }
  }
}

// ---------------------------------------------------------------------------
// Fused edge layer-1: assemble [s[row] | s[col] | (edge_s) | Gram(f_cat)] in
// LDS (f16, zero-padded to Kp) for MT edges, then WMMA GEMM with edge1
// weights and fused SiLU -> H f16 [E,HID].
__global__ __launch_bounds__(256)
void k_edge1(const float* __restrict__ f, int Fn,
             const float* __restrict__ s,
             const int* __restrict__ row, const int* __restrict__ col,
             const float* __restrict__ xpos,           // !=null: edge_f=x[r]-x[c]
             const float* __restrict__ edgeF, int Ef,
             const float* __restrict__ edgeS, int Es,  // 0 or HID
             const _Float16* __restrict__ WT, const float* __restrict__ bias,
             _Float16* __restrict__ H, int E, int Kp) {
  int fm = 2 * Fn + Ef;
  extern __shared__ char smraw[];
  _Float16* At = (_Float16*)smraw;                 // MT x stride
  int stride = Kp + LDS_PAD;
  float* fcat = (float*)(At + MT * stride);        // MT x 28
  int tid = threadIdx.x;
  int ebase = blockIdx.x * MT;

  unsigned* z32 = (unsigned*)At;
  for (int i = tid; i < MT * (stride >> 1); i += 256) z32[i] = 0;
  __syncthreads();

  int g = tid >> 3, li = tid & 7;                  // edge slot / lane-in-slot
  int e = ebase + g;
  bool ok = e < E;
  int r = 0, c = 0;
  if (ok) { r = row[e]; c = col[e]; }
  if (ok) {
    for (int idx = li; idx < 3 * fm; idx += 8) {
      int i = idx / fm, j = idx - i * fm;
      float v;
      if (j < Fn)           v = f[((size_t)r * 3 + i) * Fn + j];
      else if (j < 2 * Fn)  v = f[((size_t)c * 3 + i) * Fn + (j - Fn)];
      else {
        int jj = j - 2 * Fn;
        v = xpos ? (xpos[r * 3 + i] - xpos[c * 3 + i])
                 : edgeF[((size_t)e * 3 + i) * Ef + jj];
      }
      fcat[g * 28 + idx] = v;
    }
    for (int k = li; k < HID; k += 8) {
      At[g * stride + k]       = (_Float16)s[(size_t)r * HID + k];
      At[g * stride + HID + k] = (_Float16)s[(size_t)c * HID + k];
      if (Es) At[g * stride + 2 * HID + k] = (_Float16)edgeS[(size_t)e * HID + k];
    }
  }
  __syncthreads();
  if (ok) {
    int sb = 2 * HID + Es;
    for (int idx = li; idx < fm * fm; idx += 8) {
      int a = idx / fm, b = idx - a * fm;
      float acc = 0.f;
#pragma unroll
      for (int i = 0; i < 3; ++i)
        acc += fcat[g * 28 + i * fm + a] * fcat[g * 28 + i * fm + b];
      At[g * stride + sb + idx] = (_Float16)acc;
    }
  }
  __syncthreads();

  // GEMM: N = HID, 8 waves x 16 cols, 2 M sub-tiles
  int wave = tid >> 5, lane = tid & 31;
  int nb = wave * 16;
  v8f acc0 = {}, acc1 = {};
  const _Float16* WTn = WT + (size_t)nb * Kp;
  for (int kb = 0; kb < Kp; kb += 32) {
    v16h b  = load_b_frag(WTn, Kp, kb, lane);
    v16h a0 = load_a_frag(At, stride, kb, lane);
    v16h a1 = load_a_frag(At + 16 * stride, stride, kb, lane);
    acc0 = __builtin_amdgcn_wmma_f32_16x16x32_f16(false, a0, false, b,
                                                  (short)0, acc0, false, false);
    acc1 = __builtin_amdgcn_wmma_f32_16x16x32_f16(false, a1, false, b,
                                                  (short)0, acc1, false, false);
  }
  int n  = nb + (lane & 15);
  int rb = (lane >> 4) << 3;
  float bv = bias[n];
#pragma unroll
  for (int t = 0; t < 2; ++t) {
#pragma unroll
    for (int q = 0; q < 8; ++q) {
      int ee = ebase + t * 16 + rb + q;
      if (ee < E) {
        float v = (t ? acc1[q] : acc0[q]) + bv;
        H[(size_t)ee * HID + n] = (_Float16)silu_f(v);
      }
    }
  }
}

// ---------------------------------------------------------------------------
// Equivariant message m_f = f_cat @ coef and masked scatter-add of m_s / m_f.
__global__ __launch_bounds__(256)
void k_msg_scatter(const float* __restrict__ f, int Fn,
                   const int* __restrict__ row, const int* __restrict__ col,
                   const float* __restrict__ xpos,
                   const float* __restrict__ edgeF, int Ef,
                   const _Float16* __restrict__ mS,
                   const float* __restrict__ coef,     // [E, 2*fm]
                   const float* __restrict__ mask,     // [E] or null
                   float* __restrict__ mF,             // [E,3,2]
                   float* __restrict__ aggS,           // [Nn,HID] or null
                   float* __restrict__ aggF,           // [Nn,6]   or null
                   int E) {
  int fm = 2 * Fn + Ef;
  int tid = threadIdx.x;
  int g = tid >> 5, lane = tid & 31;
  int e = blockIdx.x * 8 + g;
  if (e >= E) return;
  int r = row[e], c = col[e];
  float mk = mask ? mask[e] : 1.f;
  if (lane < 6) {
    int i = lane >> 1, gg = lane & 1;
    float acc = 0.f;
    for (int j = 0; j < fm; ++j) {
      float fc;
      if (j < Fn)          fc = f[((size_t)r * 3 + i) * Fn + j];
      else if (j < 2 * Fn) fc = f[((size_t)c * 3 + i) * Fn + (j - Fn)];
      else fc = xpos ? (xpos[r * 3 + i] - xpos[c * 3 + i])
                     : edgeF[((size_t)e * 3 + i) * Ef + (j - 2 * Fn)];
      acc += fc * coef[(size_t)e * (2 * fm) + j * 2 + gg];
    }
    mF[(size_t)e * 6 + i * 2 + gg] = acc;
    if (aggF) atomicAdd(&aggF[(size_t)r * 6 + i * 2 + gg], acc * mk);
  }
  if (aggS)
    for (int k = lane; k < HID; k += 32)
      atomicAdd(&aggS[(size_t)r * HID + k], (float)mS[(size_t)e * HID + k] * mk);
}

// ---------------------------------------------------------------------------
// Per-node equivariant update: f += [f | agg_f] @ ncoef
__global__ __launch_bounds__(256)
void k_node_f_update(float* __restrict__ f, const float* __restrict__ aggF,
                     const float* __restrict__ ncoef, int Nn, int Fn) {
  int n = blockIdx.x * 256 + threadIdx.x;
  if (n >= Nn) return;
  int FC = Fn + 2;
  float fc[3][6];
#pragma unroll
  for (int i = 0; i < 3; ++i) {
    for (int j = 0; j < Fn; ++j) fc[i][j] = f[((size_t)n * 3 + i) * Fn + j];
    fc[i][Fn]     = aggF[(size_t)n * 6 + i * 2 + 0];
    fc[i][Fn + 1] = aggF[(size_t)n * 6 + i * 2 + 1];
  }
  for (int i = 0; i < 3; ++i)
    for (int g = 0; g < Fn; ++g) {
      float acc = 0.f;
      for (int j = 0; j < FC; ++j)
        acc += fc[i][j] * ncoef[(size_t)n * (FC * Fn) + j * Fn + g];
      f[((size_t)n * 3 + i) * Fn + g] += acc;
    }
}

// cat f16 [M,2H] = [A(gathered?) | B]
__global__ __launch_bounds__(256)
void k_cat2(const float* __restrict__ A, const int* __restrict__ gidx,
            const float* __restrict__ B, _Float16* __restrict__ out,
            int M, int H) {
  int i = blockIdx.x * 256 + threadIdx.x;
  if (i >= M * H) return;
  int m = i / H, k = i - m * H;
  int ma = gidx ? gidx[m] : m;
  out[(size_t)m * (2 * H) + k]     = (_Float16)A[(size_t)ma * H + k];
  out[(size_t)m * (2 * H) + H + k] = (_Float16)B[(size_t)m * H + k];
}

// Object scatter: cnt, sum(xv), sum(sE)*ETA
__global__ __launch_bounds__(256)
void k_obj_accum(const float* __restrict__ x, const float* __restrict__ v,
                 const float* __restrict__ sE, const int* __restrict__ obj,
                 float* cnt, float* fO, float* sO, int N) {
  int tid = threadIdx.x;
  int g = tid >> 5, lane = tid & 31;
  int n = blockIdx.x * 8 + g;
  if (n >= N) return;
  int o = obj[n];
  if (lane == 0) atomicAdd(&cnt[o], 1.f);
  if (lane < 3) {
    atomicAdd(&fO[o * 6 + lane * 2 + 0], x[n * 3 + lane]);
    atomicAdd(&fO[o * 6 + lane * 2 + 1], v[n * 3 + lane]);
  }
  for (int k = lane; k < HID; k += 32)
    atomicAdd(&sO[(size_t)o * HID + k], sE[(size_t)n * HID + k] * ETA);
}

__global__ __launch_bounds__(256)
void k_obj_div(float* fO, const float* cnt, int O) {
  int i = blockIdx.x * 256 + threadIdx.x;
  if (i >= O * 6) return;
  fO[i] /= fmaxf(cnt[i / 6], 1.f);
}

__global__ __launch_bounds__(256)
void k_build_fp(const float* x, const float* v, const float* fO,
                const int* obj, float* fP, int N) {
  int n = blockIdx.x * 256 + threadIdx.x;
  if (n >= N) return;
  int o = obj[n];
  for (int i = 0; i < 3; ++i) {
    float xv = x[n * 3 + i], vv = v[n * 3 + i];
    fP[(size_t)n * 9 + i * 3 + 0] = xv - fO[o * 6 + i * 2 + 0];
    fP[(size_t)n * 9 + i * 3 + 1] = vv - fO[o * 6 + i * 2 + 1];
    fP[(size_t)n * 9 + i * 3 + 2] = vv;
  }
}

__global__ __launch_bounds__(256)
void k_edge_bucket(const int* row, const int* col, const int* obj,
                   const float* __restrict__ mF, const _Float16* __restrict__ mS,
                   float* cntE, float* eof, float* eos, int E) {
  int tid = threadIdx.x;
  int g = tid >> 5, lane = tid & 31;
  int e = blockIdx.x * 8 + g;
  if (e >= E) return;
  int em = obj[row[e]] * NOBJ + obj[col[e]];
  if (lane == 0) atomicAdd(&cntE[em], 1.f);
  if (lane < 6) atomicAdd(&eof[em * 6 + lane], mF[(size_t)e * 6 + lane]);
  for (int k = lane; k < HID; k += 32)
    atomicAdd(&eos[(size_t)em * HID + k], (float)mS[(size_t)e * HID + k]);
}

__global__ __launch_bounds__(256)
void k_bucket_div(float* eof, float* eos, const float* cntE, float* maskE, int P) {
  int p = blockIdx.x * 256 + threadIdx.x;
  if (p >= P) return;
  float c = cntE[p], d = fmaxf(c, 1.f);
  for (int i = 0; i < 6; ++i) eof[p * 6 + i] /= d;
  for (int k = 0; k < HID; ++k) eos[(size_t)p * HID + k] /= d;
  maskE[p] = c > 0.f ? 1.f : 0.f;
}

__global__ __launch_bounds__(256)
void k_make_eio(int* r, int* c, int P) {
  int e = blockIdx.x * 256 + threadIdx.x;
  if (e >= P) return;
  r[e] = e / NOBJ; c[e] = e % NOBJ;
}

__global__ __launch_bounds__(256)
void k_build_fobj(const float* fO, float* fObj, int O) {
  int i = blockIdx.x * 256 + threadIdx.x;
  if (i >= O * 3) return;
  fObj[i] = fO[(i / 3) * 6 + (i % 3) * 2 + 1];   // f_o[..., 1:]
}

__global__ __launch_bounds__(256)
void k_build_fp2(const float* fObj, const float* fP, const int* obj,
                 float* fP2, int N) {
  int n = blockIdx.x * 256 + threadIdx.x;
  if (n >= N) return;
  int o = obj[n];
  for (int i = 0; i < 3; ++i) {
    fP2[(size_t)n * 12 + i * 4 + 0] = fObj[o * 3 + i];
    for (int j = 0; j < 3; ++j)
      fP2[(size_t)n * 12 + i * 4 + 1 + j] = fP[(size_t)n * 9 + i * 3 + j];
  }
}

__global__ __launch_bounds__(256)
void k_extract(const float* fP2, float* out, int N) {
  int i = blockIdx.x * 256 + threadIdx.x;
  if (i >= N * 3) return;
  int n = i / 3, c = i - n * 3;
  out[i] = fP2[(size_t)n * 12 + c * 4 + 0];
}

// ---------------------------------------------------------------------------
struct MpnLayer {
  _Float16 *e1, *e2, *fc, *nf, *n1, *n2;
  const float *e1b, *e2b, *fcb, *nfb, *n1b, *n2b;
};

extern "C" void kernel_launch(void* const* d_in, const int* in_sizes, int n_in,
                              void* d_out, int out_size, void* d_ws, size_t ws_size,
                              hipStream_t stream) {
  const float* x_p  = (const float*)d_in[0];
  const float* v_p  = (const float*)d_in[1];
  const float* h_p  = (const float*)d_in[2];
  const int* ei_inner = (const int*)d_in[3];
  const int* ei_inter = (const int*)d_in[4];
  const int* obj_id   = (const int*)d_in[5];
  const int* in_row = ei_inner;            const int* in_col = ei_inner + NEDGE;
  const int* it_row = ei_inter;            const int* it_col = ei_inter + NEDGE;
  float* out = (float*)d_out;

  // ---- workspace bump allocator
  char* ws = (char*)d_ws;
  size_t off = 0;
  auto alloc = [&](size_t bytes) -> void* {
    void* p = ws + off;
    off = (off + bytes + 255) & ~(size_t)255;
    return p;
  };
  _Float16* hP16  = (_Float16*)alloc((size_t)NPART * 32 * 2);
  float* sE    = (float*)alloc((size_t)NPART * HID * 4);
  float* sLoc  = (float*)alloc((size_t)NPART * HID * 4);
  float* sWork = (float*)alloc((size_t)NPART * HID * 4);
  _Float16* sW16   = (_Float16*)alloc((size_t)NPART * HID * 2);
  _Float16* catN16 = (_Float16*)alloc((size_t)NPART * 2 * HID * 2);
  _Float16* hN16   = (_Float16*)alloc((size_t)NPART * HID * 2);
  float* aggS  = (float*)alloc((size_t)NPART * HID * 4);
  float* aggF  = (float*)alloc((size_t)NPART * 6 * 4);
  float* fP    = (float*)alloc((size_t)NPART * 9 * 4);
  float* fP2   = (float*)alloc((size_t)NPART * 12 * 4);
  float* ncoef = (float*)alloc((size_t)NPART * 24 * 4);
  _Float16* hE16 = (_Float16*)alloc((size_t)NEDGE * HID * 2);
  _Float16* mS16 = (_Float16*)alloc((size_t)NEDGE * HID * 2);
  float* coefE = (float*)alloc((size_t)NEDGE * 18 * 4);
  float* mF    = (float*)alloc((size_t)NEDGE * 6 * 4);
  float* cnt   = (float*)alloc(NOBJ * 4);
  float* fO    = (float*)alloc(NOBJ * 6 * 4);
  float* sO    = (float*)alloc((size_t)NOBJ * HID * 4);
  float* fObj  = (float*)alloc(NOBJ * 3 * 4);
  float* cntE  = (float*)alloc(NOBJ * NOBJ * 4);
  float* eofB  = (float*)alloc((size_t)NOBJ * NOBJ * 6 * 4);
  float* eosB  = (float*)alloc((size_t)NOBJ * NOBJ * HID * 4);
  float* maskE = (float*)alloc(NOBJ * NOBJ * 4);
  int* eioR = (int*)alloc(NOBJ * NOBJ * 4);
  int* eioC = (int*)alloc(NOBJ * NOBJ * 4);

  // ---- weight conversion (f32 [K,N] -> f16 [Npad][Kp])
  auto cvt = [&](int wi, int K, int N, int Kp, int Np) -> _Float16* {
    _Float16* dst = (_Float16*)alloc((size_t)Np * Kp * 2);
    int tot = Np * Kp;
    k_convert_wt<<<(tot + 255) / 256, 256, 0, stream>>>(
        (const float*)d_in[wi], dst, K, N, Kp, Np);
    return dst;
  };
  _Float16* embWT  = cvt(6, 16, HID, 32, HID);
  const float* embB = (const float*)d_in[7];
  _Float16* emb1WT = cvt(8, 2 * HID, HID, 2 * HID, HID);
  const float* emb1B = (const float*)d_in[9];
  _Float16* emb2WT = cvt(10, 2 * HID, HID, 2 * HID, HID);
  const float* emb2B = (const float*)d_in[11];

  auto mklayer = [&](int b, int din, int Kp, int fm, int Fn) -> MpnLayer {
    MpnLayer L;
    int fcN = 2 * fm, fcNp = (fcN + 15) & ~15;
    int nfN = (Fn + 2) * Fn, nfNp = (nfN + 15) & ~15;
    L.e1 = cvt(b + 0, din, HID, Kp, HID);       L.e1b = (const float*)d_in[b + 1];
    L.e2 = cvt(b + 2, HID, HID, HID, HID);      L.e2b = (const float*)d_in[b + 3];
    L.fc = cvt(b + 4, HID, fcN, HID, fcNp);     L.fcb = (const float*)d_in[b + 5];
    L.nf = cvt(b + 6, HID, nfN, HID, nfNp);     L.nfb = (const float*)d_in[b + 7];
    L.n1 = cvt(b + 8, 2 * HID, HID, 2 * HID, HID); L.n1b = (const float*)d_in[b + 9];
    L.n2 = cvt(b + 10, HID, HID, HID, HID);     L.n2b = (const float*)d_in[b + 11];
    return L;
  };
  MpnLayer Llocal[2] = { mklayer(12, 305, 320, 7, 3), mklayer(24, 305, 320, 7, 3) };
  MpnLayer Lo2p[2]   = { mklayer(36, 337, 352, 9, 4), mklayer(48, 337, 352, 9, 4) };
  MpnLayer Lobj[2]   = { mklayer(60, 400, 416, 4, 1), mklayer(72, 400, 416, 4, 1) };

  auto smemG = [&](int Kp) -> size_t { return (size_t)MT * (Kp + LDS_PAD) * 2; };

  // ---- MPN driver
  auto run_mpn = [&](float* f, int Fn, float* s,
                     const int* row, const int* col, int E,
                     const float* xpos, const float* edgeF, int Ef,
                     const float* edgeS, int Es,
                     const float* mask, int Nn, int Kp,
                     const MpnLayer* L, int nlayers, bool skipLastNode) {
    int fm = 2 * Fn + Ef;
    size_t smE = smemG(Kp) + (size_t)MT * 28 * 4;
    for (int l = 0; l < nlayers; ++l) {
      bool lastskip = skipLastNode && (l == nlayers - 1);
      k_edge1<<<(E + MT - 1) / MT, 256, smE, stream>>>(
          f, Fn, s, row, col, xpos, edgeF, Ef, edgeS, Es,
          L[l].e1, L[l].e1b, hE16, E, Kp);
      k_gemm<<<(E + MT - 1) / MT, 256, smemG(HID), stream>>>(
          hE16, HID, L[l].e2, L[l].e2b, nullptr, 0, mS16, HID, nullptr,
          E, HID, HID, 0);
      k_gemm<<<(E + MT - 1) / MT, 256, smemG(HID), stream>>>(
          mS16, HID, L[l].fc, L[l].fcb, coefE, 2 * fm, nullptr, 0, nullptr,
          E, 2 * fm, HID, 0);
      if (!lastskip) {
        hipMemsetAsync(aggS, 0, (size_t)Nn * HID * 4, stream);
        hipMemsetAsync(aggF, 0, (size_t)Nn * 6 * 4, stream);
      }
      k_msg_scatter<<<(E + 7) / 8, 256, 0, stream>>>(
          f, Fn, row, col, xpos, edgeF, Ef, mS16, coefE, mask, mF,
          lastskip ? nullptr : aggS, lastskip ? nullptr : aggF, E);
      if (!lastskip) {
        int tot = Nn * HID;
        k_cat2<<<(tot + 255) / 256, 256, 0, stream>>>(s, nullptr, aggS, catN16, Nn, HID);
        k_gemm<<<(Nn + MT - 1) / MT, 256, smemG(2 * HID), stream>>>(
            catN16, 2 * HID, L[l].n1, L[l].n1b, nullptr, 0, hN16, HID, nullptr,
            Nn, HID, 2 * HID, 1);
        k_gemm<<<(Nn + MT - 1) / MT, 256, smemG(HID), stream>>>(
            hN16, HID, L[l].n2, L[l].n2b, s, HID, sW16, HID, s,
            Nn, HID, HID, 0);
        int nco = (Fn + 2) * Fn;
        k_gemm<<<(Nn + MT - 1) / MT, 256, smemG(HID), stream>>>(
            sW16, HID, L[l].nf, L[l].nfb, ncoef, nco, nullptr, 0, nullptr,
            Nn, nco, HID, 0);
        k_node_f_update<<<(Nn + 255) / 256, 256, 0, stream>>>(f, aggF, ncoef, Nn, Fn);
      }
    }
  };

  // ==== 1. sE = emb(h_p)
  k_convert_act<<<(NPART * 32 + 255) / 256, 256, 0, stream>>>(h_p, hP16, NPART, 16, 32);
  k_gemm<<<(NPART + MT - 1) / MT, 256, smemG(32), stream>>>(
      hP16, 32, embWT, embB, sE, HID, nullptr, 0, nullptr, NPART, HID, 32, 0);

  // ==== 2. object scatter mean/sum, f_p
  hipMemsetAsync(cnt, 0, NOBJ * 4, stream);
  hipMemsetAsync(fO, 0, NOBJ * 24, stream);
  hipMemsetAsync(sO, 0, (size_t)NOBJ * HID * 4, stream);
  k_obj_accum<<<(NPART + 7) / 8, 256, 0, stream>>>(x_p, v_p, sE, obj_id, cnt, fO, sO, NPART);
  k_obj_div<<<(NOBJ * 6 + 255) / 256, 256, 0, stream>>>(fO, cnt, NOBJ);
  k_build_fp<<<(NPART + 255) / 256, 256, 0, stream>>>(x_p, v_p, fO, obj_id, fP, NPART);

  // ==== 3. sLoc = emb1([s_o[obj] | sE])
  k_cat2<<<(NPART * HID + 255) / 256, 256, 0, stream>>>(sO, obj_id, sE, catN16, NPART, HID);
  k_gemm<<<(NPART + MT - 1) / MT, 256, smemG(2 * HID), stream>>>(
      catN16, 2 * HID, emb1WT, emb1B, sLoc, HID, nullptr, 0, nullptr,
      NPART, HID, 2 * HID, 0);

  // ==== 4. local MPN (edge readout; last node update skippable)
  hipMemcpyAsync(sWork, sLoc, (size_t)NPART * HID * 4, hipMemcpyDeviceToDevice, stream);
  run_mpn(fP, 3, sWork, it_row, it_col, NEDGE, x_p, nullptr, 1,
          nullptr, 0, nullptr, NPART, 320, Llocal, 2, true);

  // ==== 5. bucket edges into object pairs (uses last-layer mF / mS16)
  hipMemsetAsync(cntE, 0, NOBJ * NOBJ * 4, stream);
  hipMemsetAsync(eofB, 0, (size_t)NOBJ * NOBJ * 24, stream);
  hipMemsetAsync(eosB, 0, (size_t)NOBJ * NOBJ * HID * 4, stream);
  k_edge_bucket<<<(NEDGE + 7) / 8, 256, 0, stream>>>(
      it_row, it_col, obj_id, mF, mS16, cntE, eofB, eosB, NEDGE);
  k_bucket_div<<<(NOBJ * NOBJ + 255) / 256, 256, 0, stream>>>(
      eofB, eosB, cntE, maskE, NOBJ * NOBJ);
  k_make_eio<<<(NOBJ * NOBJ + 255) / 256, 256, 0, stream>>>(eioR, eioC, NOBJ * NOBJ);

  // ==== 6. object MPN (f = f_o[...,1:], s = s_o, masked)
  k_build_fobj<<<(NOBJ * 3 + 255) / 256, 256, 0, stream>>>(fO, fObj, NOBJ);
  run_mpn(fObj, 1, sO, eioR, eioC, NOBJ * NOBJ, nullptr, eofB, 2,
          eosB, HID, maskE, NOBJ, 416, Lobj, 2, false);

  // ==== 7. f_p2, s_p2 = emb2([s_o_[obj] | sLoc])
  k_build_fp2<<<(NPART + 255) / 256, 256, 0, stream>>>(fObj, fP, obj_id, fP2, NPART);
  k_cat2<<<(NPART * HID + 255) / 256, 256, 0, stream>>>(sO, obj_id, sLoc, catN16, NPART, HID);
  k_gemm<<<(NPART + MT - 1) / MT, 256, smemG(2 * HID), stream>>>(
      catN16, 2 * HID, emb2WT, emb2B, sWork, HID, nullptr, 0, nullptr,
      NPART, HID, 2 * HID, 0);

  // ==== 8. o2p MPN over inner edges
  run_mpn(fP2, 4, sWork, in_row, in_col, NEDGE, x_p, nullptr, 1,
          nullptr, 0, nullptr, NPART, 352, Lo2p, 2, false);

  // ==== 9. output = f_p2[..., 0]
  k_extract<<<(NPART * 3 + 255) / 256, 256, 0, stream>>>(fP2, out, NPART);
  (void)in_sizes; (void)n_in; (void)out_size; (void)ws_size;
}
